// QuantLinear_35416300323327
// MI455X (gfx1250) — 8-bit matrix operand order repaired and checked by reading, compiled, not run on hardware
//
#include <hip/hip_runtime.h>
#include <hip/hip_bf16.h>
#include <stdint.h>

#define M_TOK 8192
#define KDIM  4096
#define NDIM  4096
#define GRP   32
#define GSIZE 128

#define A_TILE   (32 * GSIZE)
#define B_TILE   (64 * GSIZE)
#define OFF_A    0
#define OFF_B    (2 * A_TILE)
#define OFF_GSUM (OFF_B + 2 * B_TILE)
#define SMEM_BYTES (OFF_GSUM + 32 * GRP * 4)

typedef __attribute__((ext_vector_type(8))) int          v8i;
typedef __attribute__((ext_vector_type(4))) int          v4i;
typedef __attribute__((ext_vector_type(4))) unsigned int v4u;

#if defined(__has_builtin)
#if __has_builtin(__builtin_amdgcn_tensor_load_to_lds)
#define HAVE_TDM 1
#endif
#endif

__global__ void __launch_bounds__(256)
quant_kernel(const float* __restrict__ x,
             int8_t* __restrict__ qx,
             int*   __restrict__ gsum,
             float* __restrict__ xs)
{
    __shared__ float red[256];
    __shared__ int   gacc[GRP];

    const int m   = blockIdx.x;
    const int tid = threadIdx.x;

    if (tid < GRP) gacc[tid] = 0;

    const float* row = x + (size_t)m * KDIM + tid * 16;
    float v[16];
    #pragma unroll
    for (int i = 0; i < 4; ++i) {
        float4 f = ((const float4*)row)[i];
        v[i*4+0] = f.x; v[i*4+1] = f.y; v[i*4+2] = f.z; v[i*4+3] = f.w;
    }
    float amax = 0.f;
    #pragma unroll
    for (int i = 0; i < 16; ++i) amax = fmaxf(amax, fabsf(v[i]));

    red[tid] = amax;
    __syncthreads();
    #pragma unroll
    for (int s = 128; s > 0; s >>= 1) {
        if (tid < s) red[tid] = fmaxf(red[tid], red[tid + s]);
        __syncthreads();
    }
    const float sc  = red[0] / 127.0f;
    const float inv = 1.0f / fmaxf(sc, 1e-8f);

    int8_t q[16];
    int lsum = 0;
    #pragma unroll
    for (int i = 0; i < 16; ++i) {
        float r = rintf(v[i] * inv);
        r = fminf(fmaxf(r, -127.0f), 127.0f);
        int qi = (int)r;
        q[i] = (int8_t)qi;
        lsum += qi;
    }
    *(int4*)(qx + (size_t)m * KDIM + tid * 16) = *(const int4*)q;

    atomicAdd(&gacc[tid >> 3], lsum);
    __syncthreads();
    if (tid < GRP) gsum[(size_t)m * GRP + tid] = gacc[tid];
    if (tid == 0)  xs[m] = sc;
}

__global__ void __launch_bounds__(256)
pack_kernel(const int* __restrict__ qw, uint8_t* __restrict__ wp, int n4)
{
    int idx = blockIdx.x * blockDim.x + threadIdx.x;
    if (idx < n4) {
        int4 c = ((const int4*)qw)[idx];
        uchar4 p;
        p.x = (uint8_t)c.x; p.y = (uint8_t)c.y;
        p.z = (uint8_t)c.z; p.w = (uint8_t)c.w;
        ((uchar4*)wp)[idx] = p;
    }
}

#ifdef HAVE_TDM
__device__ __forceinline__ void tdm_load_tile(const void* gptr, uint32_t lds_off,
                                              uint32_t tile_w_bytes, uint32_t tile_rows,
                                              uint32_t row_stride_bytes,
                                              uint32_t tensor_w_bytes, uint32_t tensor_rows)
{
    const uint64_t ga = (uint64_t)(uintptr_t)gptr;
    v4u g0 = { 1u,
               lds_off,
               (uint32_t)ga,
               (uint32_t)((ga >> 32) & 0x1FFFFFFull)
                 | (2u << 30) };
    v8i g1 = { 0,
               (int)(tensor_w_bytes << 16),
               (int)((tensor_w_bytes >> 16) | ((tensor_rows & 0xFFFFu) << 16)),
               (int)((tensor_rows >> 16) | (tile_w_bytes << 16)),
               (int)(tile_rows & 0xFFFFu),
               (int)row_stride_bytes,
               0, 0 };
    v4i g2 = { 0, 0, 0, 0 };
    v4i g3 = { 0, 0, 0, 0 };
    v8i g4 = { 0, 0, 0, 0, 0, 0, 0, 0 };
    __builtin_amdgcn_tensor_load_to_lds(g0, g1, g2, g3, g4, 0);
}
#endif

__global__ void __launch_bounds__(256, 2)
gemm_kernel(const uint8_t* __restrict__ qx,
            const uint8_t* __restrict__ wpack,
            const int*     __restrict__ gsum,
            const float*   __restrict__ xs,
            const int*     __restrict__ zeros,
            const int*     __restrict__ qstep,
            const float*   __restrict__ base_scales,
            const float*   __restrict__ step_scales,
            float*         __restrict__ out)
{
    extern __shared__ uint8_t smem[];

    const int tid  = threadIdx.x;
    const int wave = tid >> 5;
    const int lane = tid & 31;
    const int wm   = wave >> 2;
    const int wn   = wave & 3;

    const int m_blk = blockIdx.y * 32;
    const int n_blk = blockIdx.x * 64;

    int* lds_gsum = (int*)(smem + OFF_GSUM);
    for (int i = tid; i < 32 * GRP; i += 256) {
        lds_gsum[i] = gsum[(size_t)(m_blk + (i >> 5)) * GRP + (i & 31)];
    }

    const int r  = lane & 15;
    const int lh = lane >> 4;

    const int m0 = m_blk + wm * 16;
    const int n0 = n_blk + wn * 16;
    const int n_lane = n0 + r;

    const float bs = base_scales[n_lane];
    const float ss = step_scales[n_lane];
    const int*  zrow = zeros + (size_t)n_lane * GRP;
    const int*  qrow = qstep + (size_t)n_lane * GRP;

    const uint8_t* agbl = qx    + (size_t)m_blk * KDIM;
    const uint8_t* bgbl = wpack + (size_t)n_blk * KDIM;

    float facc[8];
    #pragma unroll
    for (int i = 0; i < 8; ++i) facc[i] = 0.f;

    const int mrow_base = wm * 16;

#ifdef HAVE_TDM
    if (wave == 0) {
        tdm_load_tile(agbl, OFF_A, GSIZE, 32, KDIM, KDIM, M_TOK);
        tdm_load_tile(bgbl, OFF_B, GSIZE, 64, KDIM, KDIM, NDIM);
        __builtin_amdgcn_s_wait_tensorcnt(0);
    }
#endif
    __syncthreads();

    for (int g = 0; g < GRP; ++g) {
#ifdef HAVE_TDM
        const int buf = g & 1;
        if ((wave == 0) && (g + 1 < GRP)) {
            const int k1 = (g + 1) * GSIZE;
            tdm_load_tile(agbl + k1, OFF_A + (buf ^ 1) * A_TILE, GSIZE, 32, KDIM, KDIM, M_TOK);
            tdm_load_tile(bgbl + k1, OFF_B + (buf ^ 1) * B_TILE, GSIZE, 64, KDIM, KDIM, NDIM);
        }
#else
        const int buf = 0;
        {
            const int k0 = g * GSIZE;
            __syncthreads();
            for (int i = tid; i < (A_TILE + B_TILE) / 16; i += 256) {
                int4 v; uint32_t dst;
                if (i < A_TILE / 16) {
                    const int row = (i * 16) / GSIZE, col = (i * 16) % GSIZE;
                    v = *(const int4*)(agbl + (size_t)row * KDIM + k0 + col);
                    dst = OFF_A + i * 16;
                } else {
                    const int j = i - A_TILE / 16;
                    const int row = (j * 16) / GSIZE, col = (j * 16) % GSIZE;
                    v = *(const int4*)(bgbl + (size_t)row * KDIM + k0 + col);
                    dst = OFF_B + j * 16;
                }
                *(int4*)(smem + dst) = v;
            }
            __syncthreads();
            if (g + 1 < GRP) {
                __builtin_prefetch(agbl + (size_t)(m0 - m_blk + r) * KDIM + k0 + GSIZE, 0, 3);
                __builtin_prefetch(bgbl + (size_t)(n0 - n_blk + r) * KDIM + k0 + GSIZE, 0, 3);
            }
        }
#endif
        const uint8_t* aL = smem + OFF_A + buf * A_TILE + (mrow_base + r) * GSIZE + lh * 8;
        const uint8_t* bL = smem + OFF_B + buf * B_TILE + (wn * 16 + r) * GSIZE + lh * 8;

        v8i iacc = {};
        #pragma unroll
        for (int kk = 0; kk < 2; ++kk) {
            v8i a, b;
            const uint8_t* ap = aL + kk * 64;
            ((int2*)&a)[0] = *(const int2*)(ap + 0);
            ((int2*)&a)[1] = *(const int2*)(ap + 16);
            ((int2*)&a)[2] = *(const int2*)(ap + 32);
            ((int2*)&a)[3] = *(const int2*)(ap + 48);
            const uint8_t* bp = bL + kk * 64;
            ((int2*)&b)[0] = *(const int2*)(bp + 0);
            ((int2*)&b)[1] = *(const int2*)(bp + 16);
            ((int2*)&b)[2] = *(const int2*)(bp + 32);
            ((int2*)&b)[3] = *(const int2*)(bp + 48);
            iacc = __builtin_amdgcn_wmma_i32_16x16x64_iu8(
                        true, a,  false, b,
                       iacc,  false,  false);
        }

        const int   z  = zrow[g];
        const float sc = bs + ss * (float)qrow[g];
        #pragma unroll
        for (int i = 0; i < 8; ++i) {
            const int mloc = mrow_base + i + lh * 8;
            const int gs = lds_gsum[mloc * GRP + g];
            facc[i] += sc * (float)(iacc[i] - z * gs);
        }

#ifdef HAVE_TDM
        if ((wave == 0) && (g + 1 < GRP))
            __builtin_amdgcn_s_wait_tensorcnt(0);
        __syncthreads();
#endif
    }

    #pragma unroll
    for (int i = 0; i < 8; ++i) {
        const int m = m0 + i + lh * 8;
        out[(size_t)m * NDIM + n_lane] = facc[i] * xs[m];
    }
}

extern "C" void kernel_launch(void* const* d_in, const int* in_sizes, int n_in,
                              void* d_out, int out_size, void* d_ws, size_t ws_size,
                              hipStream_t stream)
{
    const float* x           = (const float*)d_in[0];
    const int*   qweight     = (const int*)  d_in[1];
    const int*   zeros       = (const int*)  d_in[2];
    const int*   qstep       = (const int*)  d_in[3];
    const float* base_scales = (const float*)d_in[4];
    const float* step_scales = (const float*)d_in[5];
    float*       out         = (float*)d_out;

    uint8_t* ws   = (uint8_t*)d_ws;
    int8_t*  qx   = (int8_t*)ws;
    uint8_t* wp   = ws + (size_t)M_TOK * KDIM;
    int*     gsum = (int*)(ws + (size_t)M_TOK * KDIM
                              + (size_t)NDIM * KDIM);
    float*   xs   = (float*)((uint8_t*)gsum
                              + (size_t)M_TOK * GRP * sizeof(int));

    const int n4 = (NDIM * KDIM) / 4;
    pack_kernel <<<(n4 + 255) / 256, 256, 0, stream>>>(qweight, wp, n4);
    quant_kernel<<<M_TOK, 256, 0, stream>>>(x, qx, gsum, xs);

    dim3 grid(NDIM / 64, M_TOK / 32);
    gemm_kernel<<<grid, 256, SMEM_BYTES, stream>>>((const uint8_t*)qx, wp, gsum, xs,
                                                   zeros, qstep, base_scales, step_scales,
                                                   out);
}
